// SequenceRNN_73899207295071
// MI455X (gfx1250) — matrix operand order repaired and checked by reading, not compiled, not run on hardware
//
#include <hip/hip_runtime.h>
#include <hip/hip_bf16.h>

// ---------------------------------------------------------------------------
// 2-layer LSTM (B=64, T=512, I=256, H=1024) + linear head on MI455X (gfx1250).
//  * bf16 WMMA (v_wmma_f32_16x16x32_bf16) for all GEMMs, f32 accumulation.
//  * Time-parallel input GEMMs -> one huge GEMM per layer (bias fused via C).
//  * Sequential recurrent GEMM per step: 64x64 tiles + split-K x4 => 256 WGs
//    per step; Whh (8MB bf16) + h (128KB) stay resident in 192MB L2.
//  * A-slab staged in LDS per workgroup (double-buffered, async global->LDS
//    engine) -- kills the 4x redundant per-wave A loads and moves fragment
//    reads to ds_load_b128.
// ---------------------------------------------------------------------------

typedef __attribute__((ext_vector_type(8)))  __bf16 v8bf;
typedef __attribute__((ext_vector_type(16))) __bf16 v16bf;
typedef __attribute__((ext_vector_type(8)))  float  v8f;
typedef __attribute__((ext_vector_type(4)))  int    v4i;

#define B_DIM 64
#define T_DIM 512
#define I_DIM 256
#define H_DIM 1024
#define G4H   4096
#define HORIZON 24

#define KCHUNK     256      // K elements staged per LDS buffer
#define LDS_STRIDE 264      // 264 bf16 = 528B row stride (16B aligned, spreads banks)
#define SPLITK     4        // split-K factor for the per-step recurrent GEMM

#if defined(__has_builtin)
#if __has_builtin(__builtin_amdgcn_global_load_async_to_lds_b128)
#define HAVE_ASYNC_LDS 1
#endif
#if __has_builtin(__builtin_amdgcn_s_wait_asynccnt)
#define HAVE_WAIT_ASYNC_BUILTIN 1
#endif
#endif

typedef __attribute__((address_space(1))) v4i* gv4i_ptr;   // global int4*
typedef __attribute__((address_space(3))) v4i* lv4i_ptr;   // LDS int4*

__device__ __forceinline__ void wait_async_zero() {
#if defined(HAVE_ASYNC_LDS)
#if defined(HAVE_WAIT_ASYNC_BUILTIN)
    __builtin_amdgcn_s_wait_asynccnt(0);
#else
    asm volatile("s_wait_asynccnt 0x0" ::: "memory");
#endif
#endif
}

__device__ __forceinline__ float sigmf(float x) { return 1.0f / (1.0f + __expf(-x)); }

// ---------------------------------------------------------------------------
// Pack W [N,K] (f32) into WMMA B-fragment layout (16-bit B 32x16; own k order, same as the A reads):
// fragment (kt,nt) = 32 lanes x 16 bf16; lane = hi*16 + (n-nt*16); element j
// within a lane is K-contiguous: W[n][kt*32 + hi*16 + j].
// ---------------------------------------------------------------------------
__global__ __launch_bounds__(256) void pack_b_kernel(const float* __restrict__ W,
                                                     __bf16* __restrict__ Bp,
                                                     int N, int K) {
    int gid  = blockIdx.x * 256 + threadIdx.x;
    int lane = gid & 31;
    int frag = gid >> 5;
    int NT = N >> 4, KT = K >> 5;
    if (frag >= KT * NT) return;
    int kt = frag / NT, nt = frag % NT;
    int hi = lane >> 4, ln = lane & 15;
    int n = nt * 16 + ln;
    const float* src = W + (long long)n * K + kt * 32 + hi * 16;
    __bf16* dst = Bp + (long long)frag * 512 + lane * 16;
#pragma unroll
    for (int j = 0; j < 16; ++j) dst[j] = (__bf16)src[j];
}

__global__ __launch_bounds__(256) void cvt_bf16_kernel(const float* __restrict__ src,
                                                       __bf16* __restrict__ dst, int n) {
    int gid = blockIdx.x * 256 + threadIdx.x;
    if (gid < n) dst[gid] = (__bf16)src[gid];
}

__global__ __launch_bounds__(256) void zero_state_kernel(__bf16* h, float* c, float* hf) {
    int gid = blockIdx.x * 256 + threadIdx.x;
    h[gid]  = (__bf16)0.0f;
    c[gid]  = 0.0f;
    hf[gid] = 0.0f;
}

// ---------------------------------------------------------------------------
// C[M,N] partial GEMM:  C(+z slab) = A[M,K_sub] @ packed-B (+ optional bias)
//  Block: 128 threads = 4 waves; tile = 64(M) x 64(N); wave owns 64x16 strip.
//  gridDim.z = split-K factor; z-th block covers K rows [z*K/gz, (z+1)*K/gz).
//  A-slab staged to LDS in KCHUNK pieces, double buffered, async engine copy.
// ---------------------------------------------------------------------------
__global__ __launch_bounds__(128) void gemm_bf16_wmma(
    const __bf16* __restrict__ A, int lda,
    const __bf16* __restrict__ Bp,
    float* __restrict__ C, int ldc, long long cSlabStride,
    int K, int N,
    const float* __restrict__ bias1, const float* __restrict__ bias2)
{
    __shared__ __bf16 As[2][64][LDS_STRIDE];

    const int tid  = threadIdx.x;
    const int wave = tid >> 5;
    const int lane = tid & 31;
    const int hi   = lane >> 4;
    const int ln   = lane & 15;
    const int nTile = blockIdx.x * 4 + wave;
    const int nBase = nTile << 4;
    const int mBase = blockIdx.y << 6;
    const int NT    = N >> 4;
    const int col   = nBase + ln;

    const int Ksub   = K / gridDim.z;
    const int kBase0 = blockIdx.z * Ksub;
    float* Cz = C + (long long)blockIdx.z * cSlabStride;

    v8f acc[4];
    {
        float bv = bias1 ? (bias1[col] + bias2[col]) : 0.0f;
#pragma unroll
        for (int mt = 0; mt < 4; ++mt)
#pragma unroll
            for (int r = 0; r < 8; ++r) acc[mt][r] = bv;
    }

    // ---- stage one KCHUNK slab (64 rows x 256 k) : 2048 x 16B segments ----
    auto stage = [&](int buf, int kb) {
#pragma unroll
        for (int i = 0; i < 16; ++i) {
            int seg = tid + i * 128;           // 0..2047
            int r = seg >> 5;                  // row 0..63
            int c = seg & 31;                  // 16B segment within 256 k
            const __bf16* src = A + (long long)(mBase + r) * lda + kb + c * 8;
            __bf16* dst = &As[buf][r][c * 8];
#if defined(HAVE_ASYNC_LDS)
            __builtin_amdgcn_global_load_async_to_lds_b128(
                (gv4i_ptr)(void*)src, (lv4i_ptr)(void*)dst, 0, 0);
#else
            *(v8bf*)dst = *(const v8bf*)src;
#endif
        }
    };

    const int nChunks = Ksub / KCHUNK;
    stage(0, kBase0);
    wait_async_zero();
    __syncthreads();

    for (int ch = 0; ch < nChunks; ++ch) {
        if (ch + 1 < nChunks) stage((ch + 1) & 1, kBase0 + (ch + 1) * KCHUNK);

        const int buf = ch & 1;
        const int kcG0 = (kBase0 + ch * KCHUNK) >> 5;   // global 32-wide k-tile base
#pragma unroll
        for (int ks = 0; ks < KCHUNK / 32; ++ks) {
            const __bf16* bpp = Bp + ((long long)(kcG0 + ks) * NT + nTile) * 512 + lane * 16;
            v16bf bfrag = *(const v16bf*)bpp;
            const int kl = ks * 32 + hi * 16;
#pragma unroll
            for (int mt = 0; mt < 4; ++mt) {
                const __bf16* ar = &As[buf][mt * 16 + ln][kl];
                v8bf alo = *(const v8bf*)ar;
                v8bf ahi = *(const v8bf*)(ar + 8);
                v16bf afrag = __builtin_shufflevector(alo, ahi,
                    0, 1, 2, 3, 4, 5, 6, 7, 8, 9, 10, 11, 12, 13, 14, 15);
                acc[mt] = __builtin_amdgcn_wmma_f32_16x16x32_bf16(
                    false, afrag, false, bfrag, (short)0, acc[mt], false, false);
            }
        }
        if (ch + 1 < nChunks) {
            wait_async_zero();
            __syncthreads();
        }
    }

    // C/D layout: VGPR r -> row hi*8+r, lane&15 -> col
#pragma unroll
    for (int mt = 0; mt < 4; ++mt) {
        int r0 = mBase + mt * 16 + hi * 8;
#pragma unroll
        for (int r = 0; r < 8; ++r)
            Cz[(long long)(r0 + r) * ldc + col] = acc[mt][r];
    }
}

// ---------------------------------------------------------------------------
// LSTM pointwise: gates = xg[:,t,:] + sum_z partial[z];  update c, h.
// ---------------------------------------------------------------------------
__global__ __launch_bounds__(256) void lstm_elem_kernel(
    const float* __restrict__ partials,     // [SPLITK][B,4H]
    const float* __restrict__ xg,           // [B,T,4H]
    long long xgOff,                        // t*4H
    float* __restrict__ c,
    __bf16* __restrict__ h,
    float* __restrict__ hf32,
    __bf16* __restrict__ hseq,              // nullable
    long long seqOff)                       // t*H
{
    int gid = blockIdx.x * 256 + threadIdx.x;   // 0 .. B*H-1
    int b = gid >> 10;
    int j = gid & (H_DIM - 1);
    float g[4];
#pragma unroll
    for (int gate = 0; gate < 4; ++gate) {
        long long idx = (long long)b * G4H + gate * H_DIM + j;
        float s = xg[(long long)b * T_DIM * G4H + xgOff + gate * H_DIM + j];
#pragma unroll
        for (int z = 0; z < SPLITK; ++z)
            s += partials[(long long)z * B_DIM * G4H + idx];
        g[gate] = s;
    }
    float cn = sigmf(g[1]) * c[gid] + sigmf(g[0]) * tanhf(g[2]);
    float hn = sigmf(g[3]) * tanhf(cn);
    c[gid]    = cn;
    h[gid]    = (__bf16)hn;
    hf32[gid] = hn;
    if (hseq) hseq[(long long)b * T_DIM * H_DIM + seqOff + j] = (__bf16)hn;
}

__global__ __launch_bounds__(256) void head_kernel(const float* __restrict__ h,
                                                   const float* __restrict__ Wh,
                                                   const float* __restrict__ bh,
                                                   float* __restrict__ out) {
    int gid = blockIdx.x * 256 + threadIdx.x;
    if (gid >= B_DIM * HORIZON) return;
    int b = gid / HORIZON, r = gid % HORIZON;
    const float* hp = h + (long long)b * H_DIM;
    const float* wp = Wh + (long long)r * H_DIM;
    float s = bh[r];
    for (int k = 0; k < H_DIM; ++k) s += hp[k] * wp[k];
    out[gid] = s;
}

// ---------------------------------------------------------------------------
extern "C" void kernel_launch(void* const* d_in, const int* in_sizes, int n_in,
                              void* d_out, int out_size, void* d_ws, size_t ws_size,
                              hipStream_t stream) {
    const float* x     = (const float*)d_in[0];
    const float* Wih0  = (const float*)d_in[1];
    const float* Whh0  = (const float*)d_in[2];
    const float* bih0  = (const float*)d_in[3];
    const float* bhh0  = (const float*)d_in[4];
    const float* Wih1  = (const float*)d_in[5];
    const float* Whh1  = (const float*)d_in[6];
    const float* bih1  = (const float*)d_in[7];
    const float* bhh1  = (const float*)d_in[8];
    const float* Whead = (const float*)d_in[9];
    const float* bhead = (const float*)d_in[10];
    float* out = (float*)d_out;

    char* ws = (char*)d_ws;
    size_t off = 0;
    auto carve = [&](size_t bytes) -> char* {
        char* p = ws + off;
        off = (off + bytes + 255) & ~(size_t)255;
        return p;
    };
    const long long MT = (long long)B_DIM * T_DIM;                       // 32768
    float*  xg    = (float*)  carve(MT * G4H * sizeof(float));           // 512 MB
    __bf16* h1seq = (__bf16*) carve(MT * H_DIM * sizeof(__bf16));        // 64 MB
    float*  part  = (float*)  carve((size_t)SPLITK * B_DIM * G4H * sizeof(float)); // 4 MB
    __bf16* hbuf  = (__bf16*) carve((size_t)B_DIM * H_DIM * sizeof(__bf16));
    float*  cbuf  = (float*)  carve((size_t)B_DIM * H_DIM * sizeof(float));
    float*  hf32  = (float*)  carve((size_t)B_DIM * H_DIM * sizeof(float));
    __bf16* Wih0p = (__bf16*) carve((size_t)(I_DIM / 32) * (G4H / 16) * 512 * sizeof(__bf16));
    __bf16* Whh0p = (__bf16*) carve((size_t)(H_DIM / 32) * (G4H / 16) * 512 * sizeof(__bf16));
    __bf16* Wih1p = (__bf16*) carve((size_t)(H_DIM / 32) * (G4H / 16) * 512 * sizeof(__bf16));
    __bf16* Whh1p = (__bf16*) carve((size_t)(H_DIM / 32) * (G4H / 16) * 512 * sizeof(__bf16));
    __bf16* xbf   = (__bf16*) carve(MT * I_DIM * sizeof(__bf16));        // 16 MB
    (void)ws_size;

    // ---- 1) pack weights ----------------------------------------------------
    {
        int frags = (I_DIM / 32) * (G4H / 16);
        pack_b_kernel<<<frags * 32 / 256, 256, 0, stream>>>(Wih0, Wih0p, G4H, I_DIM);
        frags = (H_DIM / 32) * (G4H / 16);
        pack_b_kernel<<<frags * 32 / 256, 256, 0, stream>>>(Whh0, Whh0p, G4H, H_DIM);
        pack_b_kernel<<<frags * 32 / 256, 256, 0, stream>>>(Wih1, Wih1p, G4H, H_DIM);
        pack_b_kernel<<<frags * 32 / 256, 256, 0, stream>>>(Whh1, Whh1p, G4H, H_DIM);
    }

    // ---- 2) x -> bf16 -------------------------------------------------------
    {
        int n = (int)(MT * I_DIM);
        cvt_bf16_kernel<<<(n + 255) / 256, 256, 0, stream>>>(x, xbf, n);
    }

    const dim3 bigGrid(G4H / 64, (unsigned)(MT / 64), 1);
    const dim3 stepGrid(G4H / 64, 1, SPLITK);
    const int  stateBlocks = (B_DIM * H_DIM) / 256;
    const long long stepSlab = (long long)B_DIM * G4H;

    // ---- 3) layer 0 ---------------------------------------------------------
    gemm_bf16_wmma<<<bigGrid, 128, 0, stream>>>(xbf, I_DIM, Wih0p, xg, G4H, 0LL,
                                                I_DIM, G4H, bih0, bhh0);
    zero_state_kernel<<<stateBlocks, 256, 0, stream>>>(hbuf, cbuf, hf32);
    for (int t = 0; t < T_DIM; ++t) {
        gemm_bf16_wmma<<<stepGrid, 128, 0, stream>>>(hbuf, H_DIM, Whh0p, part, G4H,
                                                     stepSlab, H_DIM, G4H,
                                                     nullptr, nullptr);
        lstm_elem_kernel<<<stateBlocks, 256, 0, stream>>>(part, xg, (long long)t * G4H,
                                                          cbuf, hbuf, hf32,
                                                          h1seq, (long long)t * H_DIM);
    }

    // ---- 4) layer 1 ---------------------------------------------------------
    gemm_bf16_wmma<<<bigGrid, 128, 0, stream>>>(h1seq, H_DIM, Wih1p, xg, G4H, 0LL,
                                                H_DIM, G4H, bih1, bhh1);
    zero_state_kernel<<<stateBlocks, 256, 0, stream>>>(hbuf, cbuf, hf32);
    for (int t = 0; t < T_DIM; ++t) {
        gemm_bf16_wmma<<<stepGrid, 128, 0, stream>>>(hbuf, H_DIM, Whh1p, part, G4H,
                                                     stepSlab, H_DIM, G4H,
                                                     nullptr, nullptr);
        lstm_elem_kernel<<<stateBlocks, 256, 0, stream>>>(part, xg, (long long)t * G4H,
                                                          cbuf, hbuf, hf32,
                                                          nullptr, 0LL);
    }

    // ---- 5) head ------------------------------------------------------------
    head_kernel<<<(B_DIM * HORIZON + 255) / 256, 256, 0, stream>>>(hf32, Whead, bhead, out);
}
